// MultiHeadAttention_62586263437757
// MI455X (gfx1250) — compile-verified, matrix operand order checked by reading, not run on hardware
//
#include <hip/hip_runtime.h>
#include <hip/hip_bf16.h>
#include <cstdint>
#include <cstddef>
#include <cmath>

// ---------------------------------------------------------------------------
// MI455X (gfx1250) causal multi-head attention, bf16 WMMA pipeline.
// B=4, T=2048, DIM=1024, HEADS=16, HEAD_DIM=64.
// Attention computed transposed (S^T = K Q^T, O^T = V^T P^T): softmax stats
// are per-lane scalars, P^T needs no layout shuffle, causal mask only touches
// the single diagonal tile, and scores are pre-scaled by log2e so the
// hardware base-2 exp is used directly.
// ---------------------------------------------------------------------------

#define DIMC   1024
#define HEADSC 16
#define HDC    64
#define BC     4
#define TC     2048
#define MROWS  (BC * TC)          // 8192

typedef __attribute__((ext_vector_type(16))) __bf16 v16bf;
typedef __attribute__((ext_vector_type(8)))  float  v8f;

union V16BF { v16bf v; uint4 u[2]; };

__device__ __forceinline__ unsigned short f2bf_u(float f) {
    unsigned int u = __float_as_uint(f);
    u += 0x7FFFu + ((u >> 16) & 1u);          // round-to-nearest-even
    return (unsigned short)(u >> 16);
}
__device__ __forceinline__ unsigned int pk2(float a, float b) {
    return (unsigned int)f2bf_u(a) | ((unsigned int)f2bf_u(b) << 16);
}
// Fast packed f32->bf16x2 (round-to-nearest-away): one v_perm_b32.
__device__ __forceinline__ unsigned int pk2f(float a, float b) {
#if __has_builtin(__builtin_amdgcn_perm)
    return __builtin_amdgcn_perm(__float_as_uint(b) + 0x8000u,
                                 __float_as_uint(a) + 0x8000u, 0x07060302u);
#else
    return pk2(a, b);
#endif
}
// Hardware base-2 exp (v_exp_f32).
__device__ __forceinline__ float fexp2(float x) {
#if __has_builtin(__builtin_amdgcn_exp2f)
    return __builtin_amdgcn_exp2f(x);
#else
    return exp2f(x);
#endif
}

// ---------------------------------------------------------------------------
// f32 -> bf16 conversion, 4 elements per thread (all sizes are /4).
// ---------------------------------------------------------------------------
__global__ void cvt_f32_bf16(const float4* __restrict__ in,
                             uint2* __restrict__ out, int n4) {
    int i = blockIdx.x * blockDim.x + threadIdx.x;
    int stride = gridDim.x * blockDim.x;
    for (; i < n4; i += stride) {
        float4 f = in[i];
        out[i] = make_uint2(pk2(f.x, f.y), pk2(f.z, f.w));
    }
}

// ---------------------------------------------------------------------------
// GEMM: C[M,N] = A[M,K] @ W[N,K]^T, bf16 WMMA, f32 accumulation.
// One 32x64 C strip per wave: 2 A fragments x 4 B fragments = 8 WMMAs per
// 12 b128 loads each K-step (1.5 loads/WMMA).
// MODE 0: bf16 out * oscale, per-head layout [B,H,T,64]   (Q: scaled, K)
// MODE 1: bf16 out, transposed [B,H,64,T]                 (V)
// MODE 2: f32 out + bias, [M,N]                           (final projection)
// ---------------------------------------------------------------------------
template <int MODE>
__global__ void gemm_wmma(const unsigned short* __restrict__ A,
                          const unsigned short* __restrict__ W,
                          unsigned short* __restrict__ obf,
                          float* __restrict__ ofp,
                          const float* __restrict__ bias,
                          float oscale) {
    const int wid  = blockIdx.x * (blockDim.x >> 5) + (threadIdx.x >> 5);
    const int lane = threadIdx.x & 31;
    const int half = lane >> 4;
    const int l15  = lane & 15;
    const int tng = wid & 15;         // 1024/64 = 16 column groups
    const int tm  = wid >> 4;         // 256 row strips of 32
    const int m0 = tm << 5, n0 = tng << 6;

    const unsigned short* pa = A + (size_t)(m0 + l15) * DIMC + 8 * half;
    const unsigned short* pb = W + (size_t)(n0 + l15) * DIMC + 8 * half;

    v8f c00 = {}, c01 = {}, c02 = {}, c03 = {};
    v8f c10 = {}, c11 = {}, c12 = {}, c13 = {};
#pragma unroll 2
    for (int k0 = 0; k0 < DIMC; k0 += 32) {
        V16BF a0, a1, b0, b1, b2, b3;
        a0.u[0] = *(const uint4*)(pa + k0);
        a0.u[1] = *(const uint4*)(pa + k0 + 16);
        a1.u[0] = *(const uint4*)(pa + 16 * DIMC + k0);
        a1.u[1] = *(const uint4*)(pa + 16 * DIMC + k0 + 16);
        b0.u[0] = *(const uint4*)(pb + k0);
        b0.u[1] = *(const uint4*)(pb + k0 + 16);
        b1.u[0] = *(const uint4*)(pb + 16 * DIMC + k0);
        b1.u[1] = *(const uint4*)(pb + 16 * DIMC + k0 + 16);
        b2.u[0] = *(const uint4*)(pb + 32 * DIMC + k0);
        b2.u[1] = *(const uint4*)(pb + 32 * DIMC + k0 + 16);
        b3.u[0] = *(const uint4*)(pb + 48 * DIMC + k0);
        b3.u[1] = *(const uint4*)(pb + 48 * DIMC + k0 + 16);
        c00 = __builtin_amdgcn_wmma_f32_16x16x32_bf16(false, a0.v, false, b0.v, (short)0, c00, false, false);
        c01 = __builtin_amdgcn_wmma_f32_16x16x32_bf16(false, a0.v, false, b1.v, (short)0, c01, false, false);
        c02 = __builtin_amdgcn_wmma_f32_16x16x32_bf16(false, a0.v, false, b2.v, (short)0, c02, false, false);
        c03 = __builtin_amdgcn_wmma_f32_16x16x32_bf16(false, a0.v, false, b3.v, (short)0, c03, false, false);
        c10 = __builtin_amdgcn_wmma_f32_16x16x32_bf16(false, a1.v, false, b0.v, (short)0, c10, false, false);
        c11 = __builtin_amdgcn_wmma_f32_16x16x32_bf16(false, a1.v, false, b1.v, (short)0, c11, false, false);
        c12 = __builtin_amdgcn_wmma_f32_16x16x32_bf16(false, a1.v, false, b2.v, (short)0, c12, false, false);
        c13 = __builtin_amdgcn_wmma_f32_16x16x32_bf16(false, a1.v, false, b3.v, (short)0, c13, false, false);
    }

    v8f* accs[2][4] = { { &c00, &c01, &c02, &c03 }, { &c10, &c11, &c12, &c13 } };
#pragma unroll
    for (int r = 0; r < 2; ++r) {
#pragma unroll
        for (int s = 0; s < 4; ++s) {
            v8f acc = *accs[r][s];
            const int n = n0 + 16 * s + l15;
            const int mr = m0 + 16 * r;
            if (MODE == 2) {
                const float bv = bias[n];
#pragma unroll
                for (int e = 0; e < 8; ++e) {
                    const int m = mr + e + 8 * half;
                    ofp[(size_t)m * DIMC + n] = acc[e] + bv;
                }
            } else {
                const int h = n >> 6, d = n & 63;
#pragma unroll
                for (int e = 0; e < 8; ++e) {
                    const int m  = mr + e + 8 * half;
                    const int bb = m >> 11;            // / T
                    const int t  = m & (TC - 1);
                    size_t idx;
                    if (MODE == 0)
                        idx = ((size_t)(bb * HEADSC + h) * TC + t) * HDC + d;
                    else
                        idx = ((size_t)(bb * HEADSC + h) * HDC + d) * TC + t;
                    obf[idx] = f2bf_u(acc[e] * oscale);
                }
            }
        }
    }
}

// ---------------------------------------------------------------------------
// Flash attention (transposed), 16 queries per wave. Scores arrive already
// scaled by (1/sqrt(d))*log2(e), so softmax uses exp2 directly.
// ---------------------------------------------------------------------------
__global__ void flash_attn(const unsigned short* __restrict__ Q,
                           const unsigned short* __restrict__ K,
                           const unsigned short* __restrict__ Vt,
                           unsigned short* __restrict__ O) {
    const int warp = threadIdx.x >> 5;
    const int lane = threadIdx.x & 31;
    const int half = lane >> 4;
    const int l15  = lane & 15;
    const int wid  = blockIdx.x * 8 + warp;
    const int bh = wid >> 7;                   // 128 q-tiles per head
    const int q0 = (wid & 127) << 4;
    const int qidx = q0 + l15;                 // this lane's query

    const unsigned short* Qb = Q  + (size_t)bh * TC * HDC;
    const unsigned short* Kb = K  + (size_t)bh * TC * HDC;
    const unsigned short* Vb = Vt + (size_t)bh * HDC * TC;

    // Q^T B-fragments (scale pre-folded): column = query.
    V16BF bq0, bq1;
    {
        const unsigned short* p = Qb + (size_t)qidx * HDC + 8 * half;
        bq0.u[0] = *(const uint4*)(p);
        bq0.u[1] = *(const uint4*)(p + 16);
        bq1.u[0] = *(const uint4*)(p + 32);
        bq1.u[1] = *(const uint4*)(p + 48);
    }

    v8f o0 = {}, o1 = {}, o2 = {}, o3 = {};   // O^T d-tiles 0..3 (16 d each)
    float m_i = -1e30f, l_i = 0.f;

    auto tile = [&](int j, bool mask) __attribute__((always_inline)) {
        // K A-fragments: rows = keys, contiguous d.
        V16BF ak00, ak01, ak10, ak11;
        {
            const unsigned short* p0 = Kb + (size_t)(j + l15) * HDC + 8 * half;
            ak00.u[0] = *(const uint4*)(p0);
            ak00.u[1] = *(const uint4*)(p0 + 16);
            ak01.u[0] = *(const uint4*)(p0 + 32);
            ak01.u[1] = *(const uint4*)(p0 + 48);
            const unsigned short* p1 = p0 + 16 * HDC;
            ak10.u[0] = *(const uint4*)(p1);
            ak10.u[1] = *(const uint4*)(p1 + 16);
            ak11.u[0] = *(const uint4*)(p1 + 32);
            ak11.u[1] = *(const uint4*)(p1 + 48);
        }
        v8f st0 = {}, st1 = {};
        st0 = __builtin_amdgcn_wmma_f32_16x16x32_bf16(false, ak00.v, false, bq0.v, (short)0, st0, false, false);
        st0 = __builtin_amdgcn_wmma_f32_16x16x32_bf16(false, ak01.v, false, bq1.v, (short)0, st0, false, false);
        st1 = __builtin_amdgcn_wmma_f32_16x16x32_bf16(false, ak10.v, false, bq0.v, (short)0, st1, false, false);
        st1 = __builtin_amdgcn_wmma_f32_16x16x32_bf16(false, ak11.v, false, bq1.v, (short)0, st1, false, false);

        float xs[16];
        if (!mask) {
#pragma unroll
            for (int e = 0; e < 8; ++e) { xs[e] = st0[e]; xs[e + 8] = st1[e]; }
        } else {
#pragma unroll
            for (int e = 0; e < 8; ++e) {
                const int k0i = j + e + 8 * half;
                xs[e]     = (k0i <= qidx)      ? st0[e] : -1e30f;
                xs[e + 8] = (k0i + 16 <= qidx) ? st1[e] : -1e30f;
            }
        }
        float rm = xs[0];
#pragma unroll
        for (int i = 1; i < 16; ++i) rm = fmaxf(rm, xs[i]);
        rm = fmaxf(rm, __shfl_xor(rm, 16, 32));          // merge half-waves
        const float mn   = fmaxf(m_i, rm);
        const float corr = fexp2(m_i - mn);              // base-2 domain
        m_i = mn;

        float p[16], ps = 0.f;
#pragma unroll
        for (int i = 0; i < 16; ++i) { p[i] = fexp2(xs[i] - mn); ps += p[i]; }
        ps += __shfl_xor(ps, 16, 32);
        l_i = l_i * corr + ps;

        // P^T D-layout == B-fragment layout for O^T = V^T @ P^T. Just pack.
        V16BF bp;
        bp.u[0] = make_uint4(pk2f(p[0], p[1]),   pk2f(p[2], p[3]),
                             pk2f(p[4], p[5]),   pk2f(p[6], p[7]));
        bp.u[1] = make_uint4(pk2f(p[8], p[9]),   pk2f(p[10], p[11]),
                             pk2f(p[12], p[13]), pk2f(p[14], p[15]));

#pragma unroll
        for (int e = 0; e < 8; ++e) {
            o0[e] *= corr; o1[e] *= corr; o2[e] *= corr; o3[e] *= corr;
        }

        // V^T A-fragments: rows = d, contiguous keys.
        V16BF av;
        const unsigned short* pv = Vb + (size_t)l15 * TC + j + 8 * half;
        av.u[0] = *(const uint4*)(pv);
        av.u[1] = *(const uint4*)(pv + 16);
        o0 = __builtin_amdgcn_wmma_f32_16x16x32_bf16(false, av.v, false, bp.v, (short)0, o0, false, false);
        pv += 16 * TC;
        av.u[0] = *(const uint4*)(pv);
        av.u[1] = *(const uint4*)(pv + 16);
        o1 = __builtin_amdgcn_wmma_f32_16x16x32_bf16(false, av.v, false, bp.v, (short)0, o1, false, false);
        pv += 16 * TC;
        av.u[0] = *(const uint4*)(pv);
        av.u[1] = *(const uint4*)(pv + 16);
        o2 = __builtin_amdgcn_wmma_f32_16x16x32_bf16(false, av.v, false, bp.v, (short)0, o2, false, false);
        pv += 16 * TC;
        av.u[0] = *(const uint4*)(pv);
        av.u[1] = *(const uint4*)(pv + 16);
        o3 = __builtin_amdgcn_wmma_f32_16x16x32_bf16(false, av.v, false, bp.v, (short)0, o3, false, false);
    };

    const int jt_main = (q0 + 1) >> 5;         // fully-unmasked tiles
    for (int jt = 0; jt < jt_main; ++jt) tile(jt << 5, false);
    tile(jt_main << 5, true);                  // the single diagonal tile

    // Epilogue: lane owns query qidx; O^T element e = d (16t + 8*half + e),
    // consecutive in the [B*T, DIM] bf16 output -> vectorized b128 stores.
    const float inv = 1.0f / l_i;
    const int bb = bh >> 4, h = bh & 15;
    unsigned short* orow =
        O + (size_t)(bb * TC + qidx) * DIMC + h * HDC + 8 * half;
    *(uint4*)(orow +  0) = make_uint4(pk2f(o0[0]*inv, o0[1]*inv), pk2f(o0[2]*inv, o0[3]*inv),
                                      pk2f(o0[4]*inv, o0[5]*inv), pk2f(o0[6]*inv, o0[7]*inv));
    *(uint4*)(orow + 16) = make_uint4(pk2f(o1[0]*inv, o1[1]*inv), pk2f(o1[2]*inv, o1[3]*inv),
                                      pk2f(o1[4]*inv, o1[5]*inv), pk2f(o1[6]*inv, o1[7]*inv));
    *(uint4*)(orow + 32) = make_uint4(pk2f(o2[0]*inv, o2[1]*inv), pk2f(o2[2]*inv, o2[3]*inv),
                                      pk2f(o2[4]*inv, o2[5]*inv), pk2f(o2[6]*inv, o2[7]*inv));
    *(uint4*)(orow + 48) = make_uint4(pk2f(o3[0]*inv, o3[1]*inv), pk2f(o3[2]*inv, o3[3]*inv),
                                      pk2f(o3[4]*inv, o3[5]*inv), pk2f(o3[6]*inv, o3[7]*inv));
}

// ---------------------------------------------------------------------------
// Host-side launch.
// ---------------------------------------------------------------------------
extern "C" void kernel_launch(void* const* d_in, const int* in_sizes, int n_in,
                              void* d_out, int out_size, void* d_ws, size_t ws_size,
                              hipStream_t stream) {
    (void)in_sizes; (void)n_in; (void)out_size; (void)ws_size;

    const float* x  = (const float*)d_in[0];
    const float* Wq = (const float*)d_in[1];
    const float* Wk = (const float*)d_in[2];
    const float* Wv = (const float*)d_in[3];
    const float* Wp = (const float*)d_in[4];
    const float* bp = (const float*)d_in[5];
    float* out = (float*)d_out;

    const size_t MB = (size_t)1 << 20;
    uint8_t* ws = (uint8_t*)d_ws;
    unsigned short* xbf   = (unsigned short*)(ws);              // 16 MiB
    unsigned short* wqb   = (unsigned short*)(ws + 16 * MB);    //  2 MiB
    unsigned short* wkb   = (unsigned short*)(ws + 18 * MB);
    unsigned short* wvb   = (unsigned short*)(ws + 20 * MB);
    unsigned short* wpb   = (unsigned short*)(ws + 22 * MB);
    unsigned short* Qbf   = (unsigned short*)(ws + 24 * MB);    // 16 MiB
    unsigned short* Kbf   = (unsigned short*)(ws + 40 * MB);    // 16 MiB
    unsigned short* Vtbf  = (unsigned short*)(ws + 56 * MB);    // 16 MiB
    unsigned short* attnO = (unsigned short*)(ws + 72 * MB);    // 16 MiB

    const int nX4 = MROWS * DIMC / 4;     // 2,097,152 float4
    const int nW4 = DIMC * DIMC / 4;      //   262,144 float4

    cvt_f32_bf16<<<2048, 256, 0, stream>>>((const float4*)x,  (uint2*)xbf, nX4);
    cvt_f32_bf16<<<1024, 256, 0, stream>>>((const float4*)Wq, (uint2*)wqb, nW4);
    cvt_f32_bf16<<<1024, 256, 0, stream>>>((const float4*)Wk, (uint2*)wkb, nW4);
    cvt_f32_bf16<<<1024, 256, 0, stream>>>((const float4*)Wv, (uint2*)wvb, nW4);
    cvt_f32_bf16<<<1024, 256, 0, stream>>>((const float4*)Wp, (uint2*)wpb, nW4);

    // Q pre-scale: (1/sqrt(64)) * log2(e) -> softmax done in base-2 domain.
    const float qscale = 0.125f * 1.4426950408889634f;

    // (M/32)*(N/64) = 256*16 = 4096 strips, 8 waves/block -> 512 blocks.
    gemm_wmma<0><<<512, 256, 0, stream>>>(xbf, wqb, Qbf,  nullptr, nullptr, qscale);
    gemm_wmma<0><<<512, 256, 0, stream>>>(xbf, wkb, Kbf,  nullptr, nullptr, 1.0f);
    gemm_wmma<1><<<512, 256, 0, stream>>>(xbf, wvb, Vtbf, nullptr, nullptr, 1.0f);

    // B*H*(T/16) = 8192 waves -> 1024 blocks.
    flash_attn<<<1024, 256, 0, stream>>>(Qbf, Kbf, Vtbf, attnO);

    // Final projection: f32 output + bias.
    gemm_wmma<2><<<512, 256, 0, stream>>>(attnO, wpb, nullptr, out, bp, 1.0f);
}
